// GAT1_17257178596041
// MI455X (gfx1250) — matrix operand order repaired and checked by reading, not compiled, not run on hardware
//
#include <hip/hip_runtime.h>
#include <cstdint>

typedef __attribute__((ext_vector_type(16))) _Float16 v16h;
typedef __attribute__((ext_vector_type(8)))  _Float16 v8h;
typedef __attribute__((ext_vector_type(8)))  float    v8f;
typedef __attribute__((ext_vector_type(4)))  int      v4i;

#define GB 8
#define GN 2048
#define GF 128
#define GU 128

// CDNA5 async memory->LDS staging (no VGPR round-trip, tracked by ASYNCcnt)
#if __has_builtin(__builtin_amdgcn_global_load_async_to_lds_b128)
#define USE_ASYNC_LDS 1
typedef __attribute__((address_space(1))) v4i g_v4i;   // global 16B vector
typedef __attribute__((address_space(3))) v4i l_v4i;   // LDS 16B vector
#endif

__device__ __forceinline__ void async_stage_fence() {
#ifdef USE_ASYNC_LDS
#if __has_builtin(__builtin_amdgcn_s_wait_asynccnt)
  __builtin_amdgcn_s_wait_asynccnt(0);
#else
  asm volatile("s_wait_asynccnt 0" ::: "memory");
#endif
#endif
}

// ---------------------------------------------------------------------------
// Kernel 1: h = x @ W_pre (per row), plus e_s = h.a_snd, e_r = h.a_rec
// ---------------------------------------------------------------------------
__global__ __launch_bounds__(128) void gat_k1_project(
    const float* __restrict__ x, const float* __restrict__ W,
    const float* __restrict__ a_snd, const float* __restrict__ a_rec,
    float* __restrict__ h, float* __restrict__ e_s, float* __restrict__ e_r) {
  __shared__ float xrow[GF];
  __shared__ float red[GU];
  const int bn = blockIdx.x;            // b*N + n
  const int u  = threadIdx.x;
  xrow[u] = x[(size_t)bn * GF + u];
  __syncthreads();
  float acc = 0.f;
#pragma unroll 8
  for (int f = 0; f < GF; ++f) acc = fmaf(xrow[f], W[f * GU + u], acc);
  h[(size_t)bn * GU + u] = acc;

  red[u] = acc * a_snd[u];
  __syncthreads();
  for (int s = 64; s > 0; s >>= 1) { if (u < s) red[u] += red[u + s]; __syncthreads(); }
  float esv = red[0];
  __syncthreads();
  red[u] = acc * a_rec[u];
  __syncthreads();
  for (int s = 64; s > 0; s >>= 1) { if (u < s) red[u] += red[u + s]; __syncthreads(); }
  if (u == 0) { e_s[bn] = esv; e_r[bn] = red[0]; }
}

// ---------------------------------------------------------------------------
// Kernel 2: per-(b, receiver r) online softmax stats over senders s.
// 256 thr = 64 columns x 4 s-partitions, merged in LDS. adj coalesced along r.
// ---------------------------------------------------------------------------
__global__ __launch_bounds__(256) void gat_k2_colstats(
    const float* __restrict__ adj, const float* __restrict__ e_s,
    const float* __restrict__ e_r, float* __restrict__ Mx, float* __restrict__ invZ) {
  const int bpB = GN / 64;
  const int b   = blockIdx.x / bpB;
  const int rt  = blockIdx.x % bpB;
  const int c   = threadIdx.x & 63;
  const int p   = threadIdx.x >> 6;
  const int r   = rt * 64 + c;
  const size_t base = (size_t)b * GN * GN;
  const float er = e_r[b * GN + r];
  float m = -3.0e38f, z = 0.f;
  const int s0 = p * (GN / 4), s1 = s0 + (GN / 4);
#pragma unroll 4
  for (int s = s0; s < s1; ++s) {
    const float av = adj[base + (size_t)s * GN + r];
    if (av > 0.f || s == r) {                   // self loops: min(1, adj + I)
      float l = e_s[b * GN + s] + er;
      l = l > 0.f ? l : 0.2f * l;               // leaky_relu 0.2
      if (l > m) { z = z * __expf(m - l) + 1.0f; m = l; }
      else       { z += __expf(l - m); }
    }
  }
  __shared__ float ms[256], zs[256];
  ms[threadIdx.x] = m; zs[threadIdx.x] = z;
  __syncthreads();
  if (p == 0) {
#pragma unroll
    for (int q = 1; q < 4; ++q) {
      const float m2 = ms[c + q * 64], z2 = zs[c + q * 64];
      const float mn = fmaxf(m, m2);
      z = z * __expf(m - mn) + z2 * __expf(m2 - mn);
      m = mn;
    }
    Mx[b * GN + r]   = m;                        // self-loop guarantees z >= 1
    invZ[b * GN + r] = 1.0f / z;
  }
}

// ---------------------------------------------------------------------------
// Kernel 2t: HsT[b][u][r] = fp16( h[b][r][u] * invZ[b][r] )  (tiled transpose,
// both global sides coalesced, so k3 can stage B with b128 loads)
// ---------------------------------------------------------------------------
__global__ __launch_bounds__(256) void gat_k2t_transpose(
    const float* __restrict__ h, const float* __restrict__ invZ,
    _Float16* __restrict__ HsT) {
  __shared__ _Float16 tile[64][72];              // pad 8 halfs vs bank conflicts
  int id = blockIdx.x;
  const int uT = id & 1; id >>= 1;               // U/64 = 2
  const int rT = id % (GN / 64);
  const int b  = id / (GN / 64);
  const int r0 = rT * 64, u0 = uT * 64;
  const int tid = threadIdx.x;
#pragma unroll
  for (int t = 0; t < 16; ++t) {
    const int idx = tid + t * 256;               // 0..4095
    const int i = idx >> 6, j = idx & 63;        // r-local, u-local
    const float v = h[((size_t)b * GN + r0 + i) * GU + u0 + j] * invZ[b * GN + r0 + i];
    tile[j][i] = (_Float16)v;
  }
  __syncthreads();
#pragma unroll
  for (int t = 0; t < 16; ++t) {
    const int idx = tid + t * 256;
    const int j = idx >> 6, i = idx & 63;        // u-local row, r-local col
    HsT[((size_t)b * GU + u0 + j) * GN + r0 + i] = tile[j][i];
  }
}

// ---------------------------------------------------------------------------
// Kernel 3: pOut[kp][b] += A[b][:, kRange] @ Hs[b][kRange, :]
// A computed on the fly from adj/e_s/e_r/Mx. 256 thr = 8 waves; block tile
// 128 s x 128 u; K split 4 ways across blocks; double-buffered async stage.
// ---------------------------------------------------------------------------
#define STILE  128
#define KT     32
#define LSTR   40      // halfs per u-row in LDS (80B: 16B-aligned, staggered)
#define KSPLIT 4
#define KRANGE (GN / KSPLIT)   // 512

__global__ __launch_bounds__(256) void gat_k3_aggregate(
    const float* __restrict__ adj, const float* __restrict__ e_s,
    const float* __restrict__ e_r, const float* __restrict__ Mx,
    const _Float16* __restrict__ HsT, float* __restrict__ pOut) {
  __shared__ alignas(16) _Float16 ldsB[2][GU * LSTR];   // 2 x 10 KB
  __shared__ float erS[2][KT], mxS[2][KT];

  int id = blockIdx.x;
  const int kp = id & (KSPLIT - 1); id >>= 2;
  const int tpb    = GN / STILE;                 // 16
  const int b      = id / tpb;
  const int blockS = (id % tpb) * STILE;
  const int kStart = kp * KRANGE;

  const int tid  = threadIdx.x;
  const int wave = tid >> 5;                     // wave32
  const int lane = tid & 31;
  const int mrow = lane & 15;
  const int half = lane >> 4;
  const int kb   = half * 8;                     // A-frag K base per lane half

  const int sRow = blockS + wave * 16 + mrow;    // global sender row
  const float es = e_s[b * GN + sRow];
  const size_t abase = (size_t)b * GN * GN + (size_t)sRow * GN;
  const _Float16* __restrict__ HsTb = HsT + (size_t)b * GU * GN;   // [u][r]
  float* __restrict__ outP = pOut + (size_t)kp * GB * GN * GU;

  auto stage = [&](int k0, int buf) {
#pragma unroll
    for (int it = 0; it < 2; ++it) {
      const int c = tid + it * 256;              // 0..511
      const int u = c >> 2, kc = c & 3;
      const _Float16* gsrc = HsTb + (size_t)u * GN + k0 + kc * 8;
      _Float16* ldst = &ldsB[buf][u * LSTR + kc * 8];
#ifdef USE_ASYNC_LDS
      // GLOBAL_LOAD_ASYNC_TO_LDS_B128: DMA 16B memory->LDS, ASYNCcnt-tracked
      __builtin_amdgcn_global_load_async_to_lds_b128(
          (g_v4i*)gsrc, (l_v4i*)ldst, 0, 0);
#else
      *reinterpret_cast<v8h*>(ldst) = *reinterpret_cast<const v8h*>(gsrc);
#endif
    }
    if (tid < KT) {
      erS[buf][tid] = e_r[b * GN + k0 + tid];
      mxS[buf][tid] = Mx [b * GN + k0 + tid];
    }
  };

  v8f acc[8] = {};
  stage(kStart, 0);
  const int NSTEP = KRANGE / KT;                 // 16

  for (int step = 0; step < NSTEP; ++step) {
    const int k0  = kStart + step * KT;
    const int buf = step & 1;
    async_stage_fence();                         // our async LDS writes landed
    __syncthreads();                             // stage(buf) visible to all
    if (step + 1 < NSTEP) stage(k0 + KT, buf ^ 1);  // DMA overlaps compute

    // ---- A fragment: 16x32 f16 attention weights on the fly ----
    // lanes 0-15: elems[0..7]=K kb..kb+7, [8..15]=K 16+kb.. (kb=8*half)
    v16h a;
#pragma unroll
    for (int c = 0; c < 2; ++c) {
      const int kloc = kb + c * 16;
      const float4* ap =
          reinterpret_cast<const float4*>(adj + abase + (size_t)(k0 + kloc));
      const float4 q0 = ap[0], q1 = ap[1];       // 32B contiguous
      const float av[8] = {q0.x, q0.y, q0.z, q0.w, q1.x, q1.y, q1.z, q1.w};
#pragma unroll
      for (int j = 0; j < 8; ++j) {
        const int r = k0 + kloc + j;
        float l = es + erS[buf][kloc + j];
        l = l > 0.f ? l : 0.2f * l;
        const float w =
            (av[j] > 0.f || sRow == r) ? __expf(l - mxS[buf][kloc + j]) : 0.f;
        a[c * 8 + j] = (_Float16)w;
      }
    }

    // ---- 8 WMMAs: same A, B tiles from LDS ----
    // B layout (32x16 f16): elems[0..7]=K kb..kb+7, [8..15]=K 16+kb.. (kb=8*half)
    const int koff = 8 * half;
#pragma unroll
    for (int ut = 0; ut < 8; ++ut) {
      const int u = ut * 16 + mrow;
      const v8h lo = *reinterpret_cast<const v8h*>(&ldsB[buf][u * LSTR + koff]);
      const v8h hi = *reinterpret_cast<const v8h*>(&ldsB[buf][u * LSTR + koff + 16]);
      v16h bb;
#pragma unroll
      for (int j = 0; j < 8; ++j) { bb[j] = lo[j]; bb[j + 8] = hi[j]; }
      acc[ut] = __builtin_amdgcn_wmma_f32_16x16x32_f16(
          false, a, false, bb, (short)0, acc[ut], false, false);
    }
  }

  // ---- writeback: D layout VGPR i -> M = i + 8*half, N = mrow ----
#pragma unroll
  for (int ut = 0; ut < 8; ++ut) {
    const int col = ut * 16 + mrow;
#pragma unroll
    for (int i = 0; i < 8; ++i) {
      const int row = blockS + wave * 16 + i + 8 * half;
      outP[((size_t)b * GN + row) * GU + col] = acc[ut][i];
    }
  }
}

// ---------------------------------------------------------------------------
// Kernel 4: out = sum of the 4 K-split partials
// ---------------------------------------------------------------------------
__global__ __launch_bounds__(256) void gat_kred(
    const float* __restrict__ p, float* __restrict__ out) {
  const int idx = blockIdx.x * 256 + threadIdx.x;    // < 2M
  const size_t S = (size_t)GB * GN * GU;
  out[idx] = (p[idx] + p[idx + S]) + (p[idx + 2 * S] + p[idx + 3 * S]);
}

// ---------------------------------------------------------------------------
extern "C" void kernel_launch(void* const* d_in, const int* in_sizes, int n_in,
                              void* d_out, int out_size, void* d_ws, size_t ws_size,
                              hipStream_t stream) {
  const float* x     = (const float*)d_in[0];   // [B,N,F]
  const float* adj   = (const float*)d_in[1];   // [B,N,N]
  const float* W_pre = (const float*)d_in[2];   // [F,U]
  const float* a_snd = (const float*)d_in[3];   // [U]
  const float* a_rec = (const float*)d_in[4];   // [U]
  float* out = (float*)d_out;                   // [B,N,U]

  // workspace carve-up (~44.3 MB)
  float* h   = (float*)d_ws;                         // B*N*U f32 (8 MB)
  float* eS  = h  + (size_t)GB * GN * GU;            // B*N
  float* eR  = eS + (size_t)GB * GN;
  float* Mx  = eR + (size_t)GB * GN;
  float* iZ  = Mx + (size_t)GB * GN;
  _Float16* HsT = (_Float16*)(iZ + (size_t)GB * GN); // B*U*N f16 (4 MB)
  float* pOut = (float*)(HsT + (size_t)GB * GU * GN);// KSPLIT*B*N*U f32 (32 MB)

  gat_k1_project<<<GB * GN, 128, 0, stream>>>(x, W_pre, a_snd, a_rec, h, eS, eR);
  gat_k2_colstats<<<GB * (GN / 64), 256, 0, stream>>>(adj, eS, eR, Mx, iZ);
  gat_k2t_transpose<<<GB * (GN / 64) * (GU / 64), 256, 0, stream>>>(h, iZ, HsT);
  gat_k3_aggregate<<<GB * (GN / STILE) * KSPLIT, 256, 0, stream>>>(adj, eS, eR, Mx, HsT, pOut);
  gat_kred<<<(GB * GN * GU) / 256, 256, 0, stream>>>(pOut, out);
}
